// LSTM_23055384445638
// MI455X (gfx1250) — matrix operand order repaired and checked by reading, not compiled, not run on hardware
//
#include <hip/hip_runtime.h>
#include <hip/hip_bf16.h>

// Problem dims (fixed by the reference)
#define BB  64
#define SS  1024
#define II  256
#define HH  512
#define OO  128
#define NWG 16          // persistent workgroups for the forward scan
#define HS  32          // h-columns per workgroup
#define NL  128         // gate rows per workgroup (4 gates * HS)

// ---------------- WMMA plumbing (CDNA5 gfx1250, wave32) ----------------
typedef __bf16 v16bf __attribute__((ext_vector_type(16)));
typedef float  v8f   __attribute__((ext_vector_type(8)));

union ABFrag { v16bf v; unsigned int u[8]; };

__device__ __forceinline__ v8f wmma_bf16(const ABFrag& a, const ABFrag& b, v8f c) {
    return __builtin_amdgcn_wmma_f32_16x16x32_bf16(false, a.v, false, b.v,
                                                   (short)0, c, false, false);
}

__device__ __forceinline__ unsigned short f2bf(float f) {
    unsigned u = __float_as_uint(f);
    unsigned r = (u + 0x7FFFu + ((u >> 16) & 1u)) >> 16;   // RNE
    return (unsigned short)r;
}
__device__ __forceinline__ float sigm(float x) { return 1.0f / (1.0f + __expf(-x)); }

// ---------------- grid barrier (persistent kernel, monotonic ticket) ----------------
__device__ __forceinline__ void grid_barrier(unsigned* arrive, unsigned* release, unsigned nwg) {
    __syncthreads();
    __threadfence();
    if (threadIdx.x == 0) {
        unsigned ticket = __hip_atomic_fetch_add(arrive, 1u, __ATOMIC_ACQ_REL,
                                                 __HIP_MEMORY_SCOPE_AGENT);
        unsigned gen = ticket / nwg + 1u;
        if ((ticket % nwg) == nwg - 1u) {
            __hip_atomic_store(release, gen, __ATOMIC_RELEASE, __HIP_MEMORY_SCOPE_AGENT);
        } else {
            while (__hip_atomic_load(release, __ATOMIC_ACQUIRE,
                                     __HIP_MEMORY_SCOPE_AGENT) < gen) {
                __builtin_amdgcn_s_sleep(2);
            }
        }
    }
    __syncthreads();
    // all lanes must see the h produced by other WGPs (only lane 0 did the
    // acquiring atomic above) -> agent-scope acquire (invalidates WGP-local caches)
    __builtin_amdgcn_fence(__ATOMIC_ACQUIRE, "agent");
}

// ---------------- LDS layout (295,424 B <= 320 KB WGP LDS) ----------------
#define OFF_WHH  0              // [128][512] bf16  = 131072 B  (resident all steps)
#define OFF_WIH  131072         // [128][256] bf16  =  65536 B  (resident all steps)
#define OFF_BIAS 196608         // [128]      f32   =    512 B
#define OFF_X    197120         // [64][256]  bf16  =  32768 B  (aliased as gbuf f32 [64][128])
#define OFF_H    229888         // [64][512]  bf16  =  65536 B  (filled by async DMA)
#define SMEM_BYTES 295424

// =====================================================================
// Forward LSTM scan: 16 persistent WGs, W-slices resident in LDS,
// h exchanged via L2 + grid barrier each step. h staging uses
// GLOBAL_LOAD_ASYNC_TO_LDS_B128 overlapped with the input-projection GEMM.
// =====================================================================
__global__ void __launch_bounds__(256, 1)
lstm_fwd_scan_kernel(const float* __restrict__ x,
                     const float* __restrict__ Wih,    // [2048][256]
                     const float* __restrict__ Whh,    // [2048][512]
                     const float* __restrict__ bias,   // [2048]
                     unsigned short* __restrict__ hb0, // bf16 [64][512] ping
                     unsigned short* __restrict__ hb1, // bf16 [64][512] pong
                     float* __restrict__ hf_final,     // f32  [64][512]
                     unsigned* arrive, unsigned* release)
{
    __shared__ __align__(16) unsigned char smem[SMEM_BYTES];
    unsigned short* whh_s  = (unsigned short*)(smem + OFF_WHH);
    unsigned short* wih_s  = (unsigned short*)(smem + OFF_WIH);
    float*          bias_s = (float*)(smem + OFF_BIAS);
    unsigned short* xs     = (unsigned short*)(smem + OFF_X);
    float*          gbuf   = (float*)(smem + OFF_X);   // alias: valid after GEMM reads x
    unsigned short* hs     = (unsigned short*)(smem + OFF_H);

    const int tid   = threadIdx.x;
    const int hbase = blockIdx.x * HS;                  // this WG's h-column slice

    // LDS byte offset of the h staging buffer (shared aperture: addr[31:0] = LDS offset)
    const unsigned hs_lds = (unsigned)(unsigned long long)(void*)hs;

    // ---- one-time: gather & convert weight slices into LDS ----
    // local gate row r (0..127): gate q = r>>5, h-col = hbase + (r&31)
    for (int e = tid; e < NL * HH; e += 256) {
        int r = e >> 9, c = e & 511;
        int gr = (r >> 5) * HH + hbase + (r & 31);
        whh_s[e] = f2bf(Whh[(size_t)gr * HH + c]);
    }
    for (int e = tid; e < NL * II; e += 256) {
        int r = e >> 8, c = e & 255;
        int gr = (r >> 5) * HH + hbase + (r & 31);
        wih_s[e] = f2bf(Wih[(size_t)gr * II + c]);
    }
    if (tid < NL) {
        int gr = (tid >> 5) * HH + hbase + (tid & 31);
        bias_s[tid] = bias[gr];
    }

    // per-thread cell state: thread owns 8 consecutive elements of c-slice
    const int eb = tid >> 2;            // batch row 0..63
    const int jb = (tid & 3) * 8;       // j offset within slice 0..31
    float creg[8];
#pragma unroll
    for (int k = 0; k < 8; ++k) creg[k] = 0.0f;

    const int wave = tid >> 5;
    const int lane = tid & 31;
    const int half = lane >> 4;
    const int l15  = lane & 15;
    const int mt   = wave & 3;          // M-tile (batch rows 16*mt..)
    const int ng   = (wave >> 2) * 4;   // first of 4 N-tiles

    __syncthreads();

    for (int t = 0; t < SS; ++t) {
        unsigned short* hwrite = (t & 1) ? hb1 : hb0;
        const unsigned short* hread = (t & 1) ? hb0 : hb1;

        // ---- kick off async DMA of h_{t-1} into LDS (bypasses VGPRs,
        //      tracked by ASYNCcnt; completes under the x-projection GEMM) ----
        if (t > 0) {
            const char* hsrc = (const char*)hread;
#pragma unroll
            for (int i = 0; i < 16; ++i) {
                int c = tid + 256 * i;                 // 16B chunk id, 4096 total
                unsigned ldst = hs_lds + (unsigned)(c * 16);
                const void* g = hsrc + (size_t)c * 16;
                asm volatile("global_load_async_to_lds_b128 %0, %1, off"
                             :: "v"(ldst), "v"(g) : "memory");
            }
        }

        // ---- stage relu(x[:,t,:]) -> LDS bf16 [64][256] (coalesced) ----
        for (int e = tid; e < BB * II; e += 256) {
            int b = e >> 8, i = e & 255;
            float v = x[(size_t)b * (SS * II) + (size_t)t * II + i];
            xs[e] = f2bf(fmaxf(v, 0.0f));
        }
        // prefetch next step's x rows into L2 (global_prefetch_b8)
        if (t + 1 < SS) {
            int b = tid & 63, seg = tid >> 6;
            __builtin_prefetch(&x[(size_t)b * (SS * II) + (size_t)(t + 1) * II + seg * 64], 0, 1);
        }
        __syncthreads();   // xs staged (h DMA may still be in flight)

        // ---- GEMM part 1: g = relu(x_t) @ Wih_slice^T  (K = 256) ----
        v8f acc[4];
#pragma unroll
        for (int j = 0; j < 4; ++j) acc[j] = (v8f){0,0,0,0,0,0,0,0};

        for (int kk = 0; kk < II / 32; ++kk) {
            ABFrag a;
            int arow = mt * 16 + l15;
#pragma unroll
            for (int v = 0; v < 8; ++v) {
                int k0 = 32 * kk + ((v >= 4) ? 16 : 0) + (half ? 8 : 0) + (v & 3) * 2;
                a.u[v] = *(const unsigned int*)&xs[arow * II + k0];
            }
#pragma unroll
            for (int j = 0; j < 4; ++j) {
                ABFrag bf;
                int brow = (ng + j) * 16 + l15;
#pragma unroll
                for (int v = 0; v < 8; ++v) {
                    int bk = 32 * kk + ((v >= 4) ? 16 : 0) + (half ? 8 : 0) + (v & 3) * 2;
                    bf.u[v] = *(const unsigned int*)&wih_s[brow * II + bk];
                }
                acc[j] = wmma_bf16(a, bf, acc[j]);
            }
        }

        // ---- GEMM part 2: += h_{t-1} @ Whh_slice^T  (K = 512; h0 == 0) ----
        if (t > 0) {
            asm volatile("s_wait_asynccnt 0x0" ::: "memory");  // my DMA chunks done
            __syncthreads();                                    // everyone's chunks done
            for (int kk = 0; kk < HH / 32; ++kk) {
                ABFrag a;
                int arow = mt * 16 + l15;
#pragma unroll
                for (int v = 0; v < 8; ++v) {
                    int k0 = 32 * kk + ((v >= 4) ? 16 : 0) + (half ? 8 : 0) + (v & 3) * 2;
                    a.u[v] = *(const unsigned int*)&hs[arow * HH + k0];
                }
#pragma unroll
                for (int j = 0; j < 4; ++j) {
                    ABFrag bf;
                    int brow = (ng + j) * 16 + l15;
#pragma unroll
                    for (int v = 0; v < 8; ++v) {
                        int bk = 32 * kk + ((v >= 4) ? 16 : 0) + (half ? 8 : 0) + (v & 3) * 2;
                        bf.u[v] = *(const unsigned int*)&whh_s[brow * HH + bk];
                    }
                    acc[j] = wmma_bf16(a, bf, acc[j]);
                }
            }
        }
        __syncthreads();   // all waves done reading xs before aliasing as gbuf

        // ---- scatter gate tiles to LDS gbuf f32 [64][128] ----
#pragma unroll
        for (int j = 0; j < 4; ++j) {
#pragma unroll
            for (int r = 0; r < 8; ++r) {
                int row = mt * 16 + half * 8 + r;
                int col = (ng + j) * 16 + l15;
                gbuf[row * NL + col] = acc[j][r];
            }
        }
        __syncthreads();

        // ---- elementwise cell update for this WG's 64x32 slice ----
#pragma unroll
        for (int k = 0; k < 8; ++k) {
            int j = jb + k;
            float gi = gbuf[eb * NL + j]        + bias_s[j];
            float gf = gbuf[eb * NL + 32 + j]   + bias_s[32 + j];
            float gg = gbuf[eb * NL + 64 + j]   + bias_s[64 + j];
            float go = gbuf[eb * NL + 96 + j]   + bias_s[96 + j];
            float iv = sigm(gi), fv = sigm(gf);
            float gv = tanhf(gg), ov = sigm(go);
            float c = fv * creg[k] + iv * gv;
            creg[k] = c;
            float h = ov * tanhf(c);
            hwrite[(size_t)eb * HH + hbase + j] = f2bf(h);
            if (t == SS - 1) hf_final[(size_t)eb * HH + hbase + j] = h;
        }

        grid_barrier(arrive, release, NWG);
    }
}

// =====================================================================
// Backward direction: only hs_b[S-1] is used == ONE cell from zero carry.
// gates_b = relu(x[:,S-1,:]) @ W_ih_b^T  (bias added in elementwise)
// =====================================================================
__global__ void __launch_bounds__(256)
bwd_gates_kernel(const float* __restrict__ x,
                 const float* __restrict__ Wih_b,   // [2048][256]
                 float* __restrict__ gates)         // [64][2048] f32
{
    __shared__ unsigned short xsb[BB * II];
    for (int e = threadIdx.x; e < BB * II; e += 256) {
        int b = e >> 8, i = e & 255;
        float v = x[(size_t)b * (SS * II) + (size_t)(SS - 1) * II + i];
        xsb[e] = f2bf(fmaxf(v, 0.0f));
    }
    __syncthreads();

    const int wave = threadIdx.x >> 5, lane = threadIdx.x & 31;
    const int half = lane >> 4, l15 = lane & 15;
    const int tile = blockIdx.x * 8 + wave;   // 0..511
    const int mt = tile & 3;                  // 4 M-tiles
    const int nt = tile >> 2;                 // 128 N-tiles

    v8f acc = (v8f){0,0,0,0,0,0,0,0};
    for (int kk = 0; kk < II / 32; ++kk) {
        ABFrag a, bf;
        int arow = mt * 16 + l15;
#pragma unroll
        for (int v = 0; v < 8; ++v) {
            int k0 = 32 * kk + ((v >= 4) ? 16 : 0) + (half ? 8 : 0) + (v & 3) * 2;
            a.u[v] = *(const unsigned int*)&xsb[arow * II + k0];
        }
        int brow = nt * 16 + l15;
#pragma unroll
        for (int v = 0; v < 8; ++v) {
            int bk = 32 * kk + ((v >= 4) ? 16 : 0) + (half ? 8 : 0) + (v & 3) * 2;
            float w0 = Wih_b[(size_t)brow * II + bk];
            float w1 = Wih_b[(size_t)brow * II + bk + 1];
            bf.u[v] = (unsigned)f2bf(w0) | ((unsigned)f2bf(w1) << 16);
        }
        acc = wmma_bf16(a, bf, acc);
    }
#pragma unroll
    for (int r = 0; r < 8; ++r) {
        int row = mt * 16 + half * 8 + r;
        int col = nt * 16 + l15;
        gates[(size_t)row * (4 * HH) + col] = acc[r];
    }
}

__global__ void bwd_cell_kernel(const float* __restrict__ gates,
                                const float* __restrict__ b_b,
                                float* __restrict__ hb)   // [64][512]
{
    int idx = blockIdx.x * blockDim.x + threadIdx.x;
    if (idx >= BB * HH) return;
    int b = idx >> 9, j = idx & 511;
    float gi = gates[(size_t)b * 2048 + j]          + b_b[j];
    float gg = gates[(size_t)b * 2048 + 1024 + j]   + b_b[1024 + j];
    float go = gates[(size_t)b * 2048 + 1536 + j]   + b_b[1536 + j];
    float c = sigm(gi) * tanhf(gg);                 // f-gate * c0 == 0
    hb[idx] = sigm(go) * tanhf(c);
}

// out = concat(h_f, h_b) @ W_fc^T + b_fc   (tiny; keep full f32)
__global__ void fc_kernel(const float* __restrict__ hf,
                          const float* __restrict__ hb,
                          const float* __restrict__ Wfc,   // [128][1024]
                          const float* __restrict__ bfc,
                          float* __restrict__ out)         // [64][128]
{
    int idx = blockIdx.x * blockDim.x + threadIdx.x;
    if (idx >= BB * OO) return;
    int b = idx >> 7, o = idx & 127;
    const float* w = Wfc + (size_t)o * (2 * HH);
    float s = bfc[o];
    for (int k = 0; k < HH; ++k) s += hf[(size_t)b * HH + k] * w[k];
    for (int k = 0; k < HH; ++k) s += hb[(size_t)b * HH + k] * w[HH + k];
    out[idx] = s;
}

// =====================================================================
extern "C" void kernel_launch(void* const* d_in, const int* in_sizes, int n_in,
                              void* d_out, int out_size, void* d_ws, size_t ws_size,
                              hipStream_t stream) {
    (void)in_sizes; (void)n_in; (void)out_size; (void)ws_size;
    const float* x     = (const float*)d_in[0];
    const float* Wih_f = (const float*)d_in[1];
    const float* Whh_f = (const float*)d_in[2];
    const float* b_f   = (const float*)d_in[3];
    const float* Wih_b = (const float*)d_in[4];
    /* d_in[5] = W_hh_b: unused — reference only consumes hs_b[S-1] (zero carry) */
    const float* b_b   = (const float*)d_in[6];
    const float* Wfc   = (const float*)d_in[7];
    const float* bfc   = (const float*)d_in[8];

    char* ws = (char*)d_ws;
    unsigned*       arrive   = (unsigned*)ws;            // [0,256): barrier state
    unsigned*       release  = (unsigned*)(ws + 128);
    unsigned short* hb0      = (unsigned short*)(ws + 256);              // 64 KB
    unsigned short* hb1      = (unsigned short*)(ws + 256 + 65536);      // 64 KB
    float*          hf_final = (float*)(ws + 256 + 2 * 65536);           // 128 KB
    float*          gates_b  = (float*)(ws + 256 + 2 * 65536 + 131072);  // 512 KB
    float*          hbw      = (float*)(ws + 256 + 2 * 65536 + 131072 + 524288); // 128 KB

    hipMemsetAsync(ws, 0, 256, stream);   // barrier counters start at 0 every call

    lstm_fwd_scan_kernel<<<NWG, 256, 0, stream>>>(x, Wih_f, Whh_f, b_f,
                                                  hb0, hb1, hf_final, arrive, release);
    bwd_gates_kernel<<<64, 256, 0, stream>>>(x, Wih_b, gates_b);
    bwd_cell_kernel<<<(BB * HH + 255) / 256, 256, 0, stream>>>(gates_b, b_b, hbw);
    fc_kernel<<<(BB * OO + 255) / 256, 256, 0, stream>>>(hf_final, hbw, Wfc, bfc,
                                                         (float*)d_out);
}
